// LAINRDecoderSCENT_MemoryEfficient_81295140978769
// MI455X (gfx1250) — compile-verified, matrix operand order checked by reading, not run on hardware
//
#include <hip/hip_runtime.h>
#include <hip/hip_bf16.h>

// ---------------------------------------------------------------------------
// LAINR decoder forward on gfx1250 (MI455X).
// All dense matmuls use v_wmma_f32_16x16x32_f16 (f16 in / f32 accum),
// staged through LDS with fp32->f16 conversion. Everything else is cheap
// VALU glue (LN, flash-attention over 256 tokens, elementwise, N=3 output).
// ---------------------------------------------------------------------------

typedef __attribute__((ext_vector_type(8)))  _Float16 v8h;
typedef __attribute__((ext_vector_type(16))) _Float16 v16h;
typedef __attribute__((ext_vector_type(8)))  float    v8f;

#define NQ    9216      // 96*96 queries (batch-shared grid)
#define MB    18432     // B * NQ
#define HID   512
#define MULT  2048
#define CTX   32
#define MTOK  256
#define FEAT2 64        // 2*NFEAT
#define HD    128       // HEADS*DH
#define CHUNK 4608      // row chunk for the MULT-wide intermediate

// ---------------------------------------------------------------------------
// Tiled WMMA GEMM: C(M,N) = epi(A(M,K) @ W(K,N) + bias [, + res])
//   BM=128 BN=64 BK=32, 256 threads = 8 waves, wave tile 32x32 (2x2 WMMA).
// EPI: 0 = none, 1 = relu, 2 = gelu(tanh), 3 = +residual
// ---------------------------------------------------------------------------
template<int EPI>
__global__ __launch_bounds__(256)
void gemm_wmma_f16(const float* __restrict__ A, const float* __restrict__ W,
                   const float* __restrict__ bias, const float* res,
                   float* C, int M, int N, int K)
{
    __shared__ __align__(16) _Float16 As[128][40];  // [m][k], stride 40 halves (80B)
    __shared__ __align__(16) _Float16 Bs[64][40];   // [n][k] (W tile transposed)

    const int tid   = threadIdx.x;
    const int lane  = tid & 31;
    const int wid   = tid >> 5;
    const int tileM = blockIdx.y * 128;
    const int tileN = blockIdx.x * 64;
    const int waveM = (wid & 3) * 32;
    const int waveN = (wid >> 2) * 32;

    v8f acc[2][2] = {};

    const int kbase = (lane >= 16) ? 8 : 0;
    const int mlane = lane & 15;

    for (int kc = 0; kc < K; kc += 32) {
        // ---- stage A tile (128x32) fp32 -> f16 -------------------------------
        {
            const int row = tid >> 1;
            const int k0  = (tid & 1) * 16;
            const float* src = A + (size_t)(tileM + row) * K + kc + k0;
            if (kc + 32 < K) __builtin_prefetch(src + 32, 0, 1);   // global_prefetch_b8
            _Float16 tmp[16];
            #pragma unroll
            for (int i = 0; i < 16; ++i) tmp[i] = (_Float16)src[i];
            #pragma unroll
            for (int i = 0; i < 16; ++i) As[row][k0 + i] = tmp[i];
        }
        // ---- stage W tile (32x64) fp32 -> f16, transposed to [n][k] ----------
        {
            const int k  = tid >> 3;
            const int n0 = (tid & 7) * 8;
            const float* src = W + (size_t)(kc + k) * N + tileN + n0;
            if (kc + 32 < K) __builtin_prefetch(src + (size_t)32 * N, 0, 1);
            #pragma unroll
            for (int i = 0; i < 8; ++i) Bs[n0 + i][k] = (_Float16)src[i];
        }
        __syncthreads();

        #pragma unroll
        for (int mi = 0; mi < 2; ++mi) {
            v16h a;
            {
                const _Float16* p = &As[waveM + mi * 16 + mlane][kbase];
                v8h lo = *(const v8h*)p;
                v8h hi = *(const v8h*)(p + 16);
                #pragma unroll
                for (int i = 0; i < 8; ++i) { a[i] = lo[i]; a[i + 8] = hi[i]; }
            }
            #pragma unroll
            for (int ni = 0; ni < 2; ++ni) {
                v16h b;
                {
                    const _Float16* p = &Bs[waveN + ni * 16 + mlane][kbase];
                    v8h lo = *(const v8h*)p;
                    v8h hi = *(const v8h*)(p + 16);
                    #pragma unroll
                    for (int i = 0; i < 8; ++i) { b[i] = lo[i]; b[i + 8] = hi[i]; }
                }
                acc[mi][ni] = __builtin_amdgcn_wmma_f32_16x16x32_f16(
                    false, a, false, b, (short)0, acc[mi][ni], false, false);
            }
        }
        __syncthreads();
    }

    // ---- epilogue: D layout => col = N-lane, rows = r + 8*(lane>=16) --------
    #pragma unroll
    for (int mi = 0; mi < 2; ++mi) {
        #pragma unroll
        for (int ni = 0; ni < 2; ++ni) {
            const int col   = tileN + waveN + ni * 16 + mlane;
            const int rbase = tileM + waveM + mi * 16 + ((lane >= 16) ? 8 : 0);
            const float bv  = bias ? bias[col] : 0.0f;
            #pragma unroll
            for (int r = 0; r < 8; ++r) {
                const int row = rbase + r;
                float x = acc[mi][ni][r] + bv;
                if (EPI == 1) {
                    x = fmaxf(x, 0.0f);
                } else if (EPI == 2) {
                    float x3 = x * x * x;
                    x = 0.5f * x * (1.0f + tanhf(0.7978845608028654f *
                                                 (x + 0.044715f * x3)));
                } else if (EPI == 3) {
                    x += res[(size_t)row * N + col];
                }
                C[(size_t)row * N + col] = x;
            }
        }
    }
}

// ---------------------------------------------------------------------------
// LayerNorm over 512-wide rows: one wave per row, 16 elems/lane, shfl reduce.
// ---------------------------------------------------------------------------
__global__ __launch_bounds__(256)
void ln512_kernel(const float* __restrict__ x, const float* __restrict__ g,
                  const float* __restrict__ b, float* __restrict__ y, int M)
{
    const int lane = threadIdx.x & 31;
    const int row  = blockIdx.x * 8 + (threadIdx.x >> 5);
    if (row >= M) return;
    const float* xr = x + (size_t)row * HID;
    float vals[16], s = 0.f, ss = 0.f;
    #pragma unroll
    for (int i = 0; i < 16; ++i) {
        float v = xr[lane + i * 32];
        vals[i] = v; s += v; ss += v * v;
    }
    #pragma unroll
    for (int off = 16; off >= 1; off >>= 1) {
        s  += __shfl_xor(s,  off, 32);
        ss += __shfl_xor(ss, off, 32);
    }
    const float mean = s * (1.0f / HID);
    const float var  = ss * (1.0f / HID) - mean * mean;
    const float inv  = rsqrtf(var + 1e-5f);
    float* yr = y + (size_t)row * HID;
    #pragma unroll
    for (int i = 0; i < 16; ++i) {
        const int c = lane + i * 32;
        yr[c] = (vals[i] - mean) * inv * g[c] + b[c];
    }
}

// LayerNorm over CTX=32-wide rows (tokens): one wave per row, 1 elem/lane.
__global__ __launch_bounds__(256)
void ln_tokens_kernel(const float* __restrict__ tok, const float* __restrict__ g,
                      const float* __restrict__ b, float* __restrict__ out)
{
    const int lane = threadIdx.x & 31;
    const int row  = blockIdx.x * 8 + (threadIdx.x >> 5);
    if (row >= 2 * MTOK) return;
    float v = tok[(size_t)row * CTX + lane];
    float s = v, ss = v * v;
    #pragma unroll
    for (int off = 16; off >= 1; off >>= 1) {
        s  += __shfl_xor(s,  off, 32);
        ss += __shfl_xor(ss, off, 32);
    }
    const float mean = s * (1.0f / CTX);
    const float var  = ss * (1.0f / CTX) - mean * mean;
    out[(size_t)row * CTX + lane] = (v - mean) * rsqrtf(var + 1e-5f) * g[lane] + b[lane];
}

// ---------------------------------------------------------------------------
// Fourier features + query->token time coordinate.
// ---------------------------------------------------------------------------
__device__ __forceinline__ void gfe_one(float x, float y, const float* B,
                                        float* enc, int i)
{
    const float twopi = 6.283185307179586f;
    #pragma unroll 4
    for (int f = 0; f < 32; ++f) {
        float proj = twopi * (x * B[f * 2] + y * B[f * 2 + 1]);
        enc[(size_t)i * FEAT2 + f]      = __sinf(proj);
        enc[(size_t)i * FEAT2 + 32 + f] = __cosf(proj);
    }
}

__global__ void gfe_kernel(const float* __restrict__ coords,
                           const float* __restrict__ Bq,
                           const float* __restrict__ B0,
                           const float* __restrict__ B1,
                           float* encq, float* enc0, float* enc1, float* tarr)
{
    const int i = blockIdx.x * blockDim.x + threadIdx.x;
    if (i >= NQ) return;
    const float x = coords[(size_t)i * 2 + 0];   // coords[0] (batch-shared)
    const float y = coords[(size_t)i * 2 + 1];
    gfe_one(x, y, Bq, encq, i);
    gfe_one(x, y, B0, enc0, i);
    gfe_one(x, y, B1, enc1, i);
    int r = (int)(x * 96.0f); r = min(max(r, 0), 95);
    int c = (int)(y * 96.0f); c = min(max(c, 0), 95);
    tarr[i] = (float)(r * 96 + c) * (1.0f / (float)NQ);
}

// k/v projection: (2*256, 32) @ (32, 128); thread per output element.
__global__ void kv_kernel(const float* __restrict__ cn,
                          const float* __restrict__ wk,
                          const float* __restrict__ wv,
                          float* __restrict__ kbuf, float* __restrict__ vbuf)
{
    const int id = blockIdx.x * blockDim.x + threadIdx.x;
    if (id >= 2 * MTOK * HD) return;
    const int col = id & (HD - 1);
    const int row = id >> 7;
    const float* cr = cn + (size_t)row * CTX;
    float ak = 0.f, av = 0.f;
    #pragma unroll
    for (int kk = 0; kk < CTX; ++kk) {
        const float c = cr[kk];
        ak += c * wk[kk * HD + col];
        av += c * wv[kk * HD + col];
    }
    kbuf[id] = ak;
    vbuf[id] = av;
}

// Flash-style attention: 1 thread per (b, query), 2 heads, 256 tokens.
__global__ __launch_bounds__(256)
void attn_kernel(const float* __restrict__ q, const float* __restrict__ kbuf,
                 const float* __restrict__ vbuf, const float* __restrict__ tarr,
                 float* __restrict__ o)
{
    const int id = blockIdx.x * blockDim.x + threadIdx.x;
    if (id >= MB) return;
    const int b  = id / NQ;
    const int i  = id - b * NQ;
    const float ti = tarr[i];
    for (int h = 0; h < 2; ++h) {
        float qv[64];
        #pragma unroll
        for (int d = 0; d < 64; ++d) qv[d] = q[(size_t)i * HD + h * 64 + d];
        float mmax = -1e30f, l = 0.f;
        float oacc[64];
        #pragma unroll
        for (int d = 0; d < 64; ++d) oacc[d] = 0.f;
        for (int j = 0; j < MTOK; ++j) {
            const float* kr = kbuf + (size_t)(b * MTOK + j) * HD + h * 64;
            float s = 0.f;
            #pragma unroll
            for (int d = 0; d < 64; ++d) s += qv[d] * kr[d];
            const float pos = ((float)j + 0.5f) * (1.0f / MTOK);
            const float dd  = ti - pos;
            s = s * 0.125f - 10.0f * dd * dd;       // DH^-0.5 + rel-dist bias
            const float nm   = fmaxf(mmax, s);
            const float corr = __expf(mmax - nm);
            const float p    = __expf(s - nm);
            l = l * corr + p;
            const float* vr = vbuf + (size_t)(b * MTOK + j) * HD + h * 64;
            #pragma unroll
            for (int d = 0; d < 64; ++d) oacc[d] = oacc[d] * corr + p * vr[d];
            mmax = nm;
        }
        const float invl = 1.0f / l;
        #pragma unroll
        for (int d = 0; d < 64; ++d)
            o[(size_t)id * HD + h * 64 + d] = oacc[d] * invl;
    }
}

// mods = relu(h_l[row % NQ] + m_proj)
__global__ void addrelu_bcast_kernel(const float* __restrict__ hl,
                                     const float* __restrict__ mp,
                                     float* __restrict__ out)
{
    const size_t id = (size_t)blockIdx.x * blockDim.x + threadIdx.x;
    if (id >= (size_t)MB * HID) return;
    const size_t row  = id >> 9;            // /512
    const size_t col  = id & (HID - 1);
    const size_t qrow = row % NQ;
    out[id] = fmaxf(hl[qrow * HID + col] + mp[id], 0.0f);
}

__global__ void add_inplace_kernel(float* __restrict__ x,
                                   const float* __restrict__ y)
{
    const size_t id = (size_t)blockIdx.x * blockDim.x + threadIdx.x;
    if (id >= (size_t)MB * HID) return;
    x[id] += y[id];
}

// out[row, od] = sum_l ((hv_l + skip_l) @ out_w_l + out_b_l)
__global__ void out_kernel(const float* __restrict__ hv0, const float* __restrict__ hv1,
                           const float* __restrict__ sk0, const float* __restrict__ sk1,
                           const float* __restrict__ w0, const float* __restrict__ b0,
                           const float* __restrict__ w1, const float* __restrict__ b1,
                           float* __restrict__ out)
{
    const int id = blockIdx.x * blockDim.x + threadIdx.x;
    if (id >= MB * 3) return;
    const int od  = id % 3;
    const int row = id / 3;
    const int qrow = row % NQ;
    float acc = b0[od] + b1[od];
    const float* h0 = hv0 + (size_t)row * HID;
    const float* h1 = hv1 + (size_t)row * HID;
    const float* s0 = sk0 + (size_t)qrow * HID;
    const float* s1 = sk1 + (size_t)qrow * HID;
    for (int c = 0; c < HID; ++c) {
        acc += (h0[c] + s0[c]) * w0[c * 3 + od];
        acc += (h1[c] + s1[c]) * w1[c * 3 + od];
    }
    out[id] = acc;
}

// ---------------------------------------------------------------------------
// Host orchestration
// ---------------------------------------------------------------------------
extern "C" void kernel_launch(void* const* d_in, const int* in_sizes, int n_in,
                              void* d_out, int out_size, void* d_ws, size_t ws_size,
                              hipStream_t stream)
{
    (void)in_sizes; (void)n_in; (void)out_size; (void)ws_size;

    // ---- unpack inputs (setup_inputs() insertion order, depth-first) --------
    int ix = 0;
    auto F = [&](int i) -> const float* { return (const float*)d_in[i]; };
    const float* coords  = F(ix++);
    const float* tokens  = F(ix++);
    const float* Bq      = F(ix++);
    const float* Bl0     = F(ix++);
    const float* Bl1     = F(ix++);
    const float* query_w = F(ix++);
    const float* query_b = F(ix++);
    const float* ln_x_g  = F(ix++);
    const float* ln_x_b  = F(ix++);
    const float* ln_c_g  = F(ix++);
    const float* ln_c_b  = F(ix++);
    const float* wq      = F(ix++);
    const float* wk      = F(ix++);
    const float* wv      = F(ix++);
    const float* wo      = F(ix++);
    const float* bo      = F(ix++);
    struct FF { const float *g, *beta, *w1, *b1, *w2, *b2; };
    struct Branch { const float *w0, *b0; FF ff[4]; };
    auto readFF = [&]() {
        FF f; f.g = F(ix++); f.beta = F(ix++); f.w1 = F(ix++);
        f.b1 = F(ix++); f.w2 = F(ix++); f.b2 = F(ix++); return f;
    };
    auto readBranch = [&]() {
        Branch br; br.w0 = F(ix++); br.b0 = F(ix++);
        for (int i = 0; i < 4; ++i) br.ff[i] = readFF();
        return br;
    };
    Branch bw0  = readBranch(), bw1  = readBranch();
    Branch mod0 = readBranch(), mod1 = readBranch();
    FF hvff0 = readFF(), hvff1 = readFF();
    const float* skip_w[2]; const float* skip_b[2];
    for (int i = 0; i < 2; ++i) { skip_w[i] = F(ix++); skip_b[i] = F(ix++); }
    const float* out_w[2]; const float* out_b[2];
    for (int i = 0; i < 2; ++i) { out_w[i] = F(ix++); out_b[i] = F(ix++); }

    // ---- workspace bump allocator ------------------------------------------
    char* ws = (char*)d_ws;
    size_t off = 0;
    auto alloc = [&](size_t elems) -> float* {
        float* p = (float*)(ws + off);
        off += (elems * sizeof(float) + 255) & ~(size_t)255;
        return p;
    };
    float* encq  = alloc((size_t)NQ * FEAT2);
    float* enc0  = alloc((size_t)NQ * FEAT2);
    float* enc1  = alloc((size_t)NQ * FEAT2);
    float* tarr  = alloc(NQ);
    float* xq    = alloc((size_t)NQ * HID);
    float* lnbuf = alloc((size_t)MB * HID);      // LN scratch (xn, ff pre-norm)
    float* cn    = alloc((size_t)2 * MTOK * CTX);
    float* kbuf  = alloc((size_t)2 * MTOK * HD);
    float* vbuf  = alloc((size_t)2 * MTOK * HD);
    float* qbuf  = alloc((size_t)NQ * HD);
    float* obuf  = alloc((size_t)MB * HD);
    float* modv  = alloc((size_t)MB * HID);
    float* hl    = alloc((size_t)NQ * HID);
    float* mproj = alloc((size_t)MB * HID);
    float* mods0 = alloc((size_t)MB * HID);      // = hv[0]
    float* mods1 = alloc((size_t)MB * HID);      // becomes x -> hv[1]
    float* sk0   = alloc((size_t)NQ * HID);
    float* sk1   = alloc((size_t)NQ * HID);
    float* hbuf  = alloc((size_t)CHUNK * MULT);  // GELU intermediate (chunked)

    // ---- GEMM dispatch ------------------------------------------------------
    auto gemm = [&](int epi, const float* A, const float* W, const float* bias,
                    const float* res, float* C, int M, int N, int K) {
        dim3 g(N / 64, M / 128), blk(256);
        switch (epi) {
        case 0: gemm_wmma_f16<0><<<g, blk, 0, stream>>>(A, W, bias, res, C, M, N, K); break;
        case 1: gemm_wmma_f16<1><<<g, blk, 0, stream>>>(A, W, bias, res, C, M, N, K); break;
        case 2: gemm_wmma_f16<2><<<g, blk, 0, stream>>>(A, W, bias, res, C, M, N, K); break;
        default: gemm_wmma_f16<3><<<g, blk, 0, stream>>>(A, W, bias, res, C, M, N, K); break;
        }
    };
    auto ln512 = [&](const float* x, const float* g, const float* b, float* y, int M) {
        ln512_kernel<<<M / 8, 256, 0, stream>>>(x, g, b, y, M);
    };
    // x = FF(LN(x)) + x, in place; MULT intermediate chunked over rows.
    auto ff_res = [&](float* x, int M, const FF& f) {
        ln512(x, f.g, f.beta, lnbuf, M);
        for (int m0 = 0; m0 < M; m0 += CHUNK) {
            gemm(2, lnbuf + (size_t)m0 * HID, f.w1, f.b1, nullptr, hbuf, CHUNK, MULT, HID);
            gemm(3, hbuf, f.w2, f.b2, x + (size_t)m0 * HID, x + (size_t)m0 * HID,
                 CHUNK, HID, MULT);
        }
    };
    auto block_seq = [&](const float* xin, int Kin, int M, const Branch& br, float* xout) {
        gemm(1, xin, br.w0, br.b0, nullptr, xout, M, HID, Kin);
        for (int i = 0; i < 4; ++i) ff_res(xout, M, br.ff[i]);
    };

    // ---- 1. Fourier features + t --------------------------------------------
    gfe_kernel<<<(NQ + 255) / 256, 256, 0, stream>>>(coords, Bq, Bl0, Bl1,
                                                     encq, enc0, enc1, tarr);
    // ---- 2. query path: xq = relu(encq @ query_w + b) -----------------------
    gemm(1, encq, query_w, query_b, nullptr, xq, NQ, HID, FEAT2);
    // ---- 3. cross-attention -------------------------------------------------
    ln512(xq, ln_x_g, ln_x_b, lnbuf, NQ);                         // xn
    gemm(0, lnbuf, wq, nullptr, nullptr, qbuf, NQ, HD, HID);      // q (batch-shared)
    ln_tokens_kernel<<<(2 * MTOK) / 8, 256, 0, stream>>>(tokens, ln_c_g, ln_c_b, cn);
    kv_kernel<<<(2 * MTOK * HD) / 256, 256, 0, stream>>>(cn, wk, wv, kbuf, vbuf);
    attn_kernel<<<MB / 256, 256, 0, stream>>>(qbuf, kbuf, vbuf, tarr, obuf);
    gemm(0, obuf, wo, bo, nullptr, modv, MB, HID, HD);            // modv
    // ---- 4. branches --------------------------------------------------------
    const float* encs[2] = { enc0, enc1 };
    const Branch* bws[2]  = { &bw0, &bw1 };
    const Branch* modsB[2] = { &mod0, &mod1 };
    float* modsOut[2] = { mods0, mods1 };
    for (int kk = 0; kk < 2; ++kk) {
        block_seq(encs[kk], FEAT2, NQ, *bws[kk], hl);             // batch-shared
        block_seq(modv, HID, MB, *modsB[kk], mproj);              // batch-dependent
        addrelu_bcast_kernel<<<((size_t)MB * HID) / 256, 256, 0, stream>>>(
            hl, mproj, modsOut[kk]);
    }
    // ---- 5. hv chain: x = mods1 + mods0, then 2x ff_res ---------------------
    add_inplace_kernel<<<((size_t)MB * HID) / 256, 256, 0, stream>>>(mods1, mods0);
    ff_res(mods1, MB, hvff0);
    ff_res(mods1, MB, hvff1);
    // ---- 6. skips + output --------------------------------------------------
    gemm(0, enc0, skip_w[0], skip_b[0], nullptr, sk0, NQ, HID, FEAT2);
    gemm(0, enc1, skip_w[1], skip_b[1], nullptr, sk1, NQ, HID, FEAT2);
    out_kernel<<<(MB * 3 + 255) / 256, 256, 0, stream>>>(
        mods0, mods1, sk0, sk1, out_w[0], out_b[0], out_w[1], out_b[1],
        (float*)d_out);
}
